// DeformableAttention_23416161697807
// MI455X (gfx1250) — matrix operand order repaired and checked by reading, not compiled, not run on hardware
//
#include <hip/hip_runtime.h>
#include <hip/hip_bf16.h>

// ---------------------------------------------------------------------------
// Deformable attention on gfx1250 (MI455X), wave32 + WMMA f16 (f32 accum).
//   BS=8, NQ=8192, EMBED=256, NH=8, NP=4, HD=32, H=W=128, HW=16384
// Pipeline:
//   prep:      f32 weights -> f16 transposed copies (B operand layout)
//   GEMM1:     v = value @ w_val + b_val          -> f16  v[b][h][hw][d] (67MB, L2-resident)
//   GEMM2:     oa = query @ [w_off|w_attn]        -> f32  oa[bq][96] (raw, b_off and b_attn folded)
//   sample:    softmax(4) + bilinear gather of v  -> f16  t[bq][256]
//   GEMM3:     out = t @ w_out + b_out            -> f32  d_out
// ---------------------------------------------------------------------------

typedef __attribute__((ext_vector_type(16))) _Float16 v16h;
typedef __attribute__((ext_vector_type(8)))  _Float16 v8h;
typedef __attribute__((ext_vector_type(8)))  float    v8f;
typedef __attribute__((ext_vector_type(4)))  float    v4f;

#define DA_BS   8
#define DA_NQ   8192
#define DA_EMB  256
#define DA_NH   8
#define DA_NP   4
#define DA_HD   32
#define DA_H    128
#define DA_W    128
#define DA_HW   (DA_H * DA_W)      // 16384
#define DA_MVAL (DA_BS * DA_HW)    // 131072 rows for GEMM1
#define DA_MQ   (DA_BS * DA_NQ)    // 65536  rows for GEMM2/3
#define DA_NCAT 96                 // 64 offset cols + 32 attn cols

// ---------------- fragment loaders (match ISA VGPR layouts) ----------------

// 16-bit A-matrix 16x32 fragment from an fp32 row-major source (convert).
// lanes 0-15: M=lane, K={0..7,16..23}; lanes 16-31: M=lane-16, K={8..15,24..31}
__device__ __forceinline__ v16h load_a_f32(const float* __restrict__ A, int lda,
                                           int m0, int k0, int lane) {
  const int am = lane & 15;
  const int kh = (lane >> 4) << 3;            // 0 or 8
  const float* p = A + (size_t)(m0 + am) * lda + (k0 + kh);
  v4f a0 = *(const v4f*)(p);
  v4f a1 = *(const v4f*)(p + 4);
  v4f a2 = *(const v4f*)(p + 16);
  v4f a3 = *(const v4f*)(p + 20);
  v16h r;
#pragma unroll
  for (int i = 0; i < 4; ++i) {
    r[i]      = (_Float16)a0[i];
    r[i + 4]  = (_Float16)a1[i];
    r[i + 8]  = (_Float16)a2[i];
    r[i + 12] = (_Float16)a3[i];
  }
  return r;
}

// Same fragment from an f16 row-major source (no conversion, 16B loads).
__device__ __forceinline__ v16h load_a_f16(const _Float16* __restrict__ A, int lda,
                                           int m0, int k0, int lane) {
  const int am = lane & 15;
  const int kh = (lane >> 4) << 3;
  const _Float16* p = A + (size_t)(m0 + am) * lda + (k0 + kh);
  v8h lo = *(const v8h*)(p);
  v8h hi = *(const v8h*)(p + 16);
  v16h r;
#pragma unroll
  for (int i = 0; i < 8; ++i) { r[i] = lo[i]; r[i + 8] = hi[i]; }
  return r;
}

// 16-bit B-matrix 32x16 fragment from a pre-transposed f16 wT[N][K]:
// lane = N (0-15 / 16-31 mirror), K={0..7,16..23} in lanes 0-15, K={8..15,24..31} in lanes 16-31.
// Two 16-byte loads per lane.
__device__ __forceinline__ v16h load_bt(const _Float16* __restrict__ BT, int ldb,
                                        int n0, int k0, int lane) {
  const int bn = lane & 15;
  const int kh = (lane >> 4) << 3;            // 0 or 8
  const _Float16* p = BT + (size_t)(n0 + bn) * ldb + (k0 + kh);
  v8h lo = *(const v8h*)(p);
  v8h hi = *(const v8h*)(p + 16);
  v16h r;
#pragma unroll
  for (int i = 0; i < 8; ++i) { r[i] = lo[i]; r[i + 8] = hi[i]; }
  return r;
}

// ---------------- kernel 0: weight prep (transpose + f32->f16) -------------

__global__ __launch_bounds__(256) void k_prep_weights(
    const float* __restrict__ w_val, const float* __restrict__ w_off,
    const float* __restrict__ w_attn, const float* __restrict__ w_out,
    _Float16* __restrict__ w_valT,   // [256][256]
    _Float16* __restrict__ w_catT,   // [96][256]  rows 0-63: w_off^T, 64-95: w_attn^T
    _Float16* __restrict__ w_outT) { // [256][256]
  int i = blockIdx.x * blockDim.x + threadIdx.x;   // 155648 total
  if (i < 65536) {
    int n = i >> 8, k = i & 255;
    w_valT[i] = (_Float16)w_val[k * 256 + n];
  } else if (i < 131072) {
    int j = i - 65536;
    int n = j >> 8, k = j & 255;
    w_outT[j] = (_Float16)w_out[k * 256 + n];
  } else if (i < 155648) {
    int j = i - 131072;
    int n = j >> 8, k = j & 255;
    float w = (n < 64) ? w_off[k * 64 + n] : w_attn[k * 32 + (n - 64)];
    w_catT[j] = (_Float16)w;
  }
}

// ---------------- kernel 1: value projection (WMMA) ------------------------
// C[m][e] = value[m][:] @ w_val[:][e] + b_val[e], stored f16 at v[b][h][hw][d]
// block = 8 waves, wave => 16(M) x 64(N) strip; grid (1024, 4)

__global__ __launch_bounds__(256, 2) void k_value_proj(
    const float* __restrict__ value, const _Float16* __restrict__ wT,
    const float* __restrict__ bias, _Float16* __restrict__ v_out) {
  const int lane = threadIdx.x & 31;
  const int wave = threadIdx.x >> 5;
  const int m0 = blockIdx.x * 128 + wave * 16;
  const int n0 = blockIdx.y * 64;
  v8f acc[4] = {};
#pragma unroll
  for (int k0 = 0; k0 < DA_EMB; k0 += 32) {
    v16h a = load_a_f32(value, DA_EMB, m0, k0, lane);
#pragma unroll
    for (int t = 0; t < 4; ++t) {
      v16h b = load_bt(wT, DA_EMB, n0 + 16 * t, k0, lane);
      acc[t] = __builtin_amdgcn_wmma_f32_16x16x32_f16(
          false, a, false, b, (short)0, acc[t], false, false);
    }
  }
  const int mo = (lane >> 4) * 8;     // C layout: lanes 16-31 hold M+8
  const int nl = lane & 15;
#pragma unroll
  for (int t = 0; t < 4; ++t) {
    const int e = n0 + 16 * t + nl;
    const int h = e >> 5, d = e & 31;
    const float bv = bias[e];
#pragma unroll
    for (int r = 0; r < 8; ++r) {
      const int m = m0 + mo + r;
      const int bi = m >> 14;         // / HW
      const int hw = m & (DA_HW - 1);
      size_t idx = ((((size_t)bi * DA_NH + h) * DA_HW) + hw) * DA_HD + d;
      v_out[idx] = (_Float16)(acc[t][r] + bv);
    }
  }
}

// ---------------- kernel 2: offsets + attn logits (WMMA) -------------------
// oa[m][n] : n<64 -> offset (+b_off), n>=64 -> attn logit (+b_attn)
// wave => 16(M) x 32(N); grid (512, 3)

__global__ __launch_bounds__(256, 2) void k_offattn_proj(
    const float* __restrict__ query, const _Float16* __restrict__ wcatT,
    const float* __restrict__ b_off, const float* __restrict__ b_attn,
    float* __restrict__ oa) {
  const int lane = threadIdx.x & 31;
  const int wave = threadIdx.x >> 5;
  const int m0 = blockIdx.x * 128 + wave * 16;
  const int n0 = blockIdx.y * 32;
  v8f acc[2] = {};
#pragma unroll
  for (int k0 = 0; k0 < DA_EMB; k0 += 32) {
    v16h a = load_a_f32(query, DA_EMB, m0, k0, lane);
#pragma unroll
    for (int t = 0; t < 2; ++t) {
      v16h b = load_bt(wcatT, DA_EMB, n0 + 16 * t, k0, lane);
      acc[t] = __builtin_amdgcn_wmma_f32_16x16x32_f16(
          false, a, false, b, (short)0, acc[t], false, false);
    }
  }
  const int mo = (lane >> 4) * 8;
  const int nl = lane & 15;
#pragma unroll
  for (int t = 0; t < 2; ++t) {
    const int n = n0 + 16 * t + nl;
    const float bv = (n < 64) ? b_off[n] : b_attn[n - 64];
#pragma unroll
    for (int r = 0; r < 8; ++r) {
      const int m = m0 + mo + r;
      oa[(size_t)m * DA_NCAT + n] = acc[t][r] + bv;
    }
  }
}

// ---------------- kernel 3: softmax + bilinear sampling --------------------
// one wave per (b,q,h); lane = channel d (HD==32). v is f16, L2-resident.

__device__ __forceinline__ float da_corner(const _Float16* __restrict__ vh,
                                           int x, int y, float w, int lane) {
  if ((unsigned)x < (unsigned)DA_W && (unsigned)y < (unsigned)DA_H)
    return w * (float)vh[((size_t)(y * DA_W + x)) * DA_HD + lane];
  return 0.0f;
}

__global__ __launch_bounds__(256) void k_sample(
    const float* __restrict__ ref, const float* __restrict__ oa,
    const _Float16* __restrict__ v, _Float16* __restrict__ t_out) {
  const int lane = threadIdx.x & 31;
  const int wave = threadIdx.x >> 5;
  const int tuple = blockIdx.x * 8 + wave;   // < BS*NQ*NH = 524288
  const int h = tuple & (DA_NH - 1);
  const int bq = tuple >> 3;                 // b*NQ + q
  const int b = bq >> 13;                    // / NQ

  const float* oab = oa + (size_t)bq * DA_NCAT;
  const float rx = ref[(size_t)bq * 2 + 0];
  const float ry = ref[(size_t)bq * 2 + 1];

  // softmax over the 4 attention logits of this head
  const float l0 = oab[64 + h * 4 + 0];
  const float l1 = oab[64 + h * 4 + 1];
  const float l2 = oab[64 + h * 4 + 2];
  const float l3 = oab[64 + h * 4 + 3];
  const float mx = fmaxf(fmaxf(l0, l1), fmaxf(l2, l3));
  const float e0 = __expf(l0 - mx), e1 = __expf(l1 - mx);
  const float e2 = __expf(l2 - mx), e3 = __expf(l3 - mx);
  const float inv = 1.0f / (e0 + e1 + e2 + e3);
  float wsm[4] = {e0 * inv, e1 * inv, e2 * inv, e3 * inv};

  const _Float16* vh = v + (((size_t)b * DA_NH + h) * DA_HW) * DA_HD;
  float acc = 0.0f;
#pragma unroll
  for (int p = 0; p < DA_NP; ++p) {
    const float ox = oab[(h * DA_NP + p) * 2 + 0];
    const float oy = oab[(h * DA_NP + p) * 2 + 1];
    // loc = ref + off/[W,H]; px = loc*W - 0.5 (align_corners=False)
    const float px = (rx + ox * (1.0f / DA_W)) * (float)DA_W - 0.5f;
    const float py = (ry + oy * (1.0f / DA_H)) * (float)DA_H - 0.5f;
    const float fx = floorf(px), fy = floorf(py);
    const float wx = px - fx, wy = py - fy;
    const int x0 = (int)fx, y0 = (int)fy;
    const float wp = wsm[p];
    acc += da_corner(vh, x0,     y0,     (1.0f - wx) * (1.0f - wy) * wp, lane);
    acc += da_corner(vh, x0 + 1, y0,     wx * (1.0f - wy) * wp,          lane);
    acc += da_corner(vh, x0,     y0 + 1, (1.0f - wx) * wy * wp,          lane);
    acc += da_corner(vh, x0 + 1, y0 + 1, wx * wy * wp,                   lane);
  }
  t_out[(size_t)bq * DA_EMB + h * DA_HD + lane] = (_Float16)acc;
}

// ---------------- kernel 4: output projection (WMMA, f16 A) ----------------
// out[m][n] = t[m][:] @ w_out[:][n] + b_out[n]; grid (512, 4)

__global__ __launch_bounds__(256, 2) void k_out_proj(
    const _Float16* __restrict__ t_in, const _Float16* __restrict__ wT,
    const float* __restrict__ bias, float* __restrict__ out) {
  const int lane = threadIdx.x & 31;
  const int wave = threadIdx.x >> 5;
  const int m0 = blockIdx.x * 128 + wave * 16;
  const int n0 = blockIdx.y * 64;
  v8f acc[4] = {};
#pragma unroll
  for (int k0 = 0; k0 < DA_EMB; k0 += 32) {
    v16h a = load_a_f16(t_in, DA_EMB, m0, k0, lane);
#pragma unroll
    for (int t = 0; t < 4; ++t) {
      v16h b = load_bt(wT, DA_EMB, n0 + 16 * t, k0, lane);
      acc[t] = __builtin_amdgcn_wmma_f32_16x16x32_f16(
          false, a, false, b, (short)0, acc[t], false, false);
    }
  }
  const int mo = (lane >> 4) * 8;
  const int nl = lane & 15;
#pragma unroll
  for (int t = 0; t < 4; ++t) {
    const int n = n0 + 16 * t + nl;
    const float bv = bias[n];
#pragma unroll
    for (int r = 0; r < 8; ++r) {
      const int m = m0 + mo + r;
      out[(size_t)m * DA_EMB + n] = acc[t][r] + bv;
    }
  }
}

// ---------------------------- launcher -------------------------------------

extern "C" void kernel_launch(void* const* d_in, const int* in_sizes, int n_in,
                              void* d_out, int out_size, void* d_ws, size_t ws_size,
                              hipStream_t stream) {
  const float* query  = (const float*)d_in[0];   // [8,8192,256]
  const float* refpts = (const float*)d_in[1];   // [8,8192,2]
  const float* value  = (const float*)d_in[2];   // [8,16384,256]
  const float* w_off  = (const float*)d_in[3];   // [256,64]
  const float* b_off  = (const float*)d_in[4];   // [64]
  const float* w_attn = (const float*)d_in[5];   // [256,32]
  const float* b_attn = (const float*)d_in[6];   // [32]
  const float* w_val  = (const float*)d_in[7];   // [256,256]
  const float* b_val  = (const float*)d_in[8];   // [256]
  const float* w_out  = (const float*)d_in[9];   // [256,256]
  const float* b_out  = (const float*)d_in[10];  // [256]
  float* out = (float*)d_out;                    // [8,8192,256]

  // workspace carve-up (all 256B-aligned)
  char* ws = (char*)d_ws;
  _Float16* w_valT = (_Float16*)(ws + 0);                       // 131072 B
  _Float16* w_outT = (_Float16*)(ws + 131072);                  // 131072 B
  _Float16* w_catT = (_Float16*)(ws + 262144);                  //  49152 B
  _Float16* v      = (_Float16*)(ws + 311296);                  // 67108864 B
  float*    oa     = (float*)   (ws + 311296 + 67108864);       // 25165824 B
  _Float16* t_buf  = (_Float16*)(ws + 311296 + 67108864 + 25165824); // 33554432 B

  // 0) weight prep: 155648 elements
  k_prep_weights<<<dim3((155648 + 255) / 256), dim3(256), 0, stream>>>(
      w_val, w_off, w_attn, w_out, w_valT, w_catT, w_outT);

  // 1) value projection: M=131072, N=256
  k_value_proj<<<dim3(DA_MVAL / 128, DA_EMB / 64), dim3(256), 0, stream>>>(
      value, w_valT, b_val, v);

  // 2) offsets + attn logits: M=65536, N=96
  k_offattn_proj<<<dim3(DA_MQ / 128, DA_NCAT / 32), dim3(256), 0, stream>>>(
      query, w_catT, b_off, b_attn, oa);

  // 3) sampling: one wave per (b,q,h) -> 524288 waves / 8 per block
  k_sample<<<dim3((DA_BS * DA_NQ * DA_NH) / 8), dim3(256), 0, stream>>>(
      refpts, oa, v, t_buf);

  // 4) output projection: M=65536, N=256 -> d_out
  k_out_proj<<<dim3(DA_MQ / 128, DA_EMB / 64), dim3(256), 0, stream>>>(
      t_buf, w_outT, b_out, out);
}
